// Model_81286551044424
// MI455X (gfx1250) — matrix operand order repaired and checked by reading, not compiled, not run on hardware
//
#include <hip/hip_runtime.h>
#include <math.h>

// ---------------------------------------------------------------------------
// MI455X (gfx1250) tanh-RNN stack.
//
// Latency-critical recurrence h = tanh(pre_t + h @ Whh^T) is done with one
// v_wmma_f32_16x16x32_f16 per 16-batch tile per timestep:
//   A = Whh (f16, K padded 16->32, resident in VGPRs for the whole scan)
//   B = h^T (f16, lanes 16..31 = K 16..31 = zero padding)
//   C = pre_t tile (f32, 2x global_load_b128 per lane)
// D(f32) -> v_tanh_f32 -> output store; D->B relayout for the next step is a
// v_cvt_pk_rtz_f16 pack + one ds_swizzle SWAPX16 lane-half exchange (D and B
// are both column-per-lane layouts, split across lane halves).
// ---------------------------------------------------------------------------

typedef __attribute__((ext_vector_type(16))) _Float16 v16h;
typedef __attribute__((ext_vector_type(8)))  float    v8f;
typedef __attribute__((ext_vector_type(2)))  _Float16 v2h;

#define Tn 2048
#define Bn 64

// Branch-free tanh. CDNA5 has a hardware V_TANH_F32 transcendental; fall back
// to a clamped exp/rcp formulation (still branch-free) if the builtin is
// missing on this toolchain.
__device__ __forceinline__ float fast_tanh(float x) {
#if __has_builtin(__builtin_amdgcn_tanhf)
  return __builtin_amdgcn_tanhf(x);
#else
  x = fminf(fmaxf(x, -9.0f), 9.0f);           // tanh saturates; avoid inf/inf
  float e = __expf(2.0f * x);                 // v_exp_f32 path
  return (e - 1.0f) * __builtin_amdgcn_rcpf(e + 1.0f);
#endif
}

// Convert a D-layout f32 accumulator tile (rows j per-vgpr, column b per-lane)
// into the B-matrix f16 layout for the next WMMA (K=0..15 in lanes 0..15,
// K=16..31 zero in lanes 16..31).
__device__ __forceinline__ v16h d_to_b(v8f d, bool lowhalf) {
  int p[4], q[4];
#pragma unroll
  for (int i = 0; i < 4; ++i) {
    auto t = __builtin_amdgcn_cvt_pkrtz(d[2 * i], d[2 * i + 1]);
    p[i] = __builtin_bit_cast(int, t);
  }
#pragma unroll
  for (int i = 0; i < 4; ++i)
    q[i] = __builtin_amdgcn_ds_swizzle(p[i], 0x401F);  // SWAPX16: lane ^= 16
  v16h b;
#pragma unroll
  for (int i = 0; i < 4; ++i) {
    int lo = lowhalf ? p[i] : 0;
    int hi = lowhalf ? q[i] : 0;
    v2h l2 = __builtin_bit_cast(v2h, lo);
    v2h h2 = __builtin_bit_cast(v2h, hi);
    b[2 * i]     = l2[0]; b[2 * i + 1]     = l2[1];
    b[8 + 2 * i] = h2[0]; b[8 + 2 * i + 1] = h2[1];
  }
  return b;
}

// Serial recurrence. grid.x = #directions (dir 1 runs time-reversed).
// Block = 128 threads = 4 waves; wave w owns batch tile [16w, 16w+16).
__global__ __launch_bounds__(128) void rnn_scan_kernel(
    const float* __restrict__ pre0, const float* __restrict__ pre1,
    const float* __restrict__ whh0, const float* __restrict__ whh1,
    const float* __restrict__ hin0, const float* __restrict__ hin1,
    float* __restrict__ hout0, float* __restrict__ hout1,
    float* __restrict__ out, long long os_t, long long os_b) {
  const int dir = blockIdx.x;
  const float* pre  = dir ? pre1 : pre0;
  const float* whh  = dir ? whh1 : whh0;
  const float* hin  = dir ? hin1 : hin0;
  float*       hout = dir ? hout1 : hout0;
  const int rev     = dir;       // direction 1 is the reverse scan
  const int out_off = dir * 16;  // concat position in bidir output

  const int lane = threadIdx.x & 31;
  const int half = lane >> 4;
  const int lc   = lane & 15;
  const int b    = ((threadIdx.x >> 5) << 4) + lc;  // global batch row

  // A-matrix: Whh[j][k] in f16, K padded 16 -> 32 with zeros.
  v16h A;
#pragma unroll
  for (int k = 0; k < 8; ++k) A[k] = half ? (_Float16)0.0f : (_Float16)whh[lc * 16 + k];
#pragma unroll
  for (int k = 8; k < 16; ++k) A[k] = half ? (_Float16)0.0f : (_Float16)whh[lc * 16 + k];

  // Initial hidden state in D layout.
  v8f h;
  if (hin != nullptr) {
    const float* hp = hin + b * 16 + half * 8;
#pragma unroll
    for (int r = 0; r < 8; ++r) h[r] = hp[r];
  } else {
#pragma unroll
    for (int r = 0; r < 8; ++r) h[r] = 0.0f;
  }
  v16h hB = d_to_b(h, half == 0);

  const long long lstep = rev ? -(long long)(Bn * 16) : (long long)(Bn * 16);
  const float* pp =
      pre + ((long long)(rev ? (Tn - 1) : 0) * Bn + b) * 16 + half * 8;
  float* op_base = out + (long long)b * os_b + out_off + half * 8;

  for (int s = 0; s < Tn; ++s) {
    const int t = rev ? (Tn - 1 - s) : s;
    v8f c;
#pragma unroll
    for (int r = 0; r < 8; ++r) c[r] = pp[r];  // 2x global_load_b128
    __builtin_prefetch(pp + lstep, 0, 1);      // global_prefetch next step
    v8f d = __builtin_amdgcn_wmma_f32_16x16x32_f16(
        false, A, false, hB, (short)0, c, false, false);
#pragma unroll
    for (int r = 0; r < 8; ++r) d[r] = fast_tanh(d[r]);
    float* op = op_base + (long long)t * os_t;
#pragma unroll
    for (int r = 0; r < 8; ++r) op[r] = d[r];  // 2x global_store_b128
    hB = d_to_b(d, half == 0);
    h  = d;
    pp += lstep;
  }
  float* hp = hout + b * 16 + half * 8;
#pragma unroll
  for (int r = 0; r < 8; ++r) hp[r] = h[r];
}

// Parallel input projection: pre[t,b,:] = inp[t,b,:] @ Wih^T + bih + bhh.
// One wave per 16-row tile of the flattened (T*B, in_sz) input.
__global__ __launch_bounds__(256) void rnn_proj_kernel(
    const float* __restrict__ inp, long long is_t, long long is_b,
    const float* __restrict__ wih, const float* __restrict__ bih,
    const float* __restrict__ bhh, float* __restrict__ pre, int in_sz) {
  const int lane = threadIdx.x & 31;
  const int half = lane >> 4;
  const int lc   = lane & 15;
  const long long tile =
      (long long)blockIdx.x * (blockDim.x >> 5) + (threadIdx.x >> 5);
  const long long row0 = tile << 4;

  // B-matrix: Wih^T -> B[k][j]; lane holds column j=lc, K range = half*16..+15.
  v16h Bm;
#pragma unroll
  for (int k = 0; k < 16; ++k) {
    const int kg = half * 16 + k;
    Bm[k] = (kg < in_sz) ? (_Float16)wih[lc * in_sz + kg] : (_Float16)0.0f;
  }
  // A-matrix: 16 input rows; lane's row = row0+lc for both halves.
  const long long row = row0 + lc;
  const long long t = row / Bn, bi = row % Bn;
  const float* ip = inp + t * is_t + bi * is_b;
  v16h Am;
#pragma unroll
  for (int k = 0; k < 8; ++k) {
    const int kg  = half * 16 + k;
    const int kg2 = half * 16 + 8 + k;
    Am[k]     = (kg  < in_sz) ? (_Float16)ip[kg]  : (_Float16)0.0f;
    Am[8 + k] = (kg2 < in_sz) ? (_Float16)ip[kg2] : (_Float16)0.0f;
  }
  const float bj = bih[lc] + bhh[lc];  // C = bias broadcast over rows
  v8f c;
#pragma unroll
  for (int r = 0; r < 8; ++r) c[r] = bj;
  v8f d = __builtin_amdgcn_wmma_f32_16x16x32_f16(
      false, Am, false, Bm, (short)0, c, false, false);
  // D: lane holds column j=lc, rows row0 + 8*half + r.
  float* op = pre + (row0 + half * 8) * 16 + lc;
#pragma unroll
  for (int r = 0; r < 8; ++r) op[(long long)r * 16] = d[r];
}

// ---------------------------------------------------------------------------
// Host-side orchestration
// ---------------------------------------------------------------------------
struct DirP { const float *wih, *whh, *bih, *bhh; };

// Per-dict flatten order assumed = insertion order: Wih, Whh, bih, bhh.
static inline DirP getp(void* const* d_in, int base) {
  return {(const float*)d_in[base + 0], (const float*)d_in[base + 1],
          (const float*)d_in[base + 2], (const float*)d_in[base + 3]};
}

static void run_stack(hipStream_t st, void* const* d_in, int pbase, int layers,
                      int dirs, int in0, const float* inp, long long it,
                      long long ib, float* bufA, float* bufB, float* preB,
                      const float* hinit, float* hfin, float* outF,
                      long long oFt, long long oFb) {
  const float* cur = inp;
  long long cit = it, cib = ib;
  int csz = in0;
  float* pre0 = preB;
  float* pre1 = preB + (long long)Tn * Bn * 16;
  for (int l = 0; l < layers; ++l) {
    const int width = dirs * 16;
    const bool last = (l == layers - 1);
    float* outb;
    long long ot, ob;
    if (last) { outb = outF; ot = oFt; ob = oFb; }
    else {
      outb = (cur == bufA) ? bufB : bufA;
      ot = (long long)Bn * width; ob = width;  // (T,B,width)
    }
    for (int d = 0; d < dirs; ++d) {
      DirP p = getp(d_in, pbase + (l * dirs + d) * 4);
      rnn_proj_kernel<<<dim3(1024), dim3(256), 0, st>>>(
          cur, cit, cib, p.wih, p.bih, p.bhh, d ? pre1 : pre0, csz);
    }
    DirP p0 = getp(d_in, pbase + (l * dirs + 0) * 4);
    DirP p1 = getp(d_in, pbase + (l * dirs + (dirs > 1 ? 1 : 0)) * 4);
    const float* h0 = hinit ? hinit + (long long)(l * dirs + 0) * Bn * 16 : nullptr;
    const float* h1 = hinit ? hinit + (long long)(l * dirs + 1) * Bn * 16 : nullptr;
    float* ho0 = hfin + (long long)(l * dirs + 0) * Bn * 16;
    float* ho1 = hfin + (long long)(l * dirs + (dirs > 1 ? 1 : 0)) * Bn * 16;
    rnn_scan_kernel<<<dim3(dirs), dim3(128), 0, st>>>(
        pre0, pre1, p0.whh, p1.whh, h0, dirs > 1 ? h1 : h0, ho0, ho1, outb, ot,
        ob);
    cur = outb; cit = ot; cib = ob; csz = width;
  }
}

extern "C" void kernel_launch(void* const* d_in, const int* in_sizes, int n_in,
                              void* d_out, int out_size, void* d_ws,
                              size_t ws_size, hipStream_t stream) {
  (void)in_sizes; (void)n_in; (void)out_size; (void)ws_size;
  const float* x = (const float*)d_in[0];  // (B,T,32)
  const float* y = (const float*)d_in[1];  // (B,T,25)

  // Workspace carve-up (floats). Total ~59 MB.
  float* w = (float*)d_ws;
  float* bufA = w; w += (long long)Tn * Bn * 32;
  float* bufB = w; w += (long long)Tn * Bn * 32;
  float* mid  = w; w += (long long)Tn * Bn * 16;  // saved stack-1 output
  float* preB = w; w += (long long)Tn * Bn * 32;  // pre for dir0 + dir1
  float* hA = w; w += 8 * Bn * 16;
  float* hB = w; w += 8 * Bn * 16;
  float* hD = w; w += 8 * Bn * 16;  // dummy sink

  float* OUT = (float*)d_out;
  const long long OS = (long long)Bn * Tn * 32;  // elems per output tensor
  // Final outputs are (B,T,32): os_t = 32, os_b = T*32.
  const long long fOt = 32, fOb = (long long)Tn * 32;

  // Flattened param index map: x,y then p00..p14, 4 arrays per layer-dir.
  const int P00 = 2, P01 = 6, P02 = 18, P03 = 50, P04 = 82;
  const int P10 = 114, P11 = 118, P12 = 130, P13 = 162;  // p14@194 unused

  // ---- x branch ----
  // p00: in (B,T,32) strided, out -> bufA (T,B,16)
  run_stack(stream, d_in, P00, 1, 1, 32, x, 32, (long long)Tn * 32, bufA, bufB,
            preB, nullptr, hD, bufA, (long long)Bn * 16, 16);
  // p01: 3 layers unidir, out -> mid (= x1)
  run_stack(stream, d_in, P01, 3, 1, 16, bufA, (long long)Bn * 16, 16, bufA,
            bufB, preB, nullptr, hD, mid, (long long)Bn * 16, 16);
  // p02/p03/p04: 4-layer bidir, chained hidden states, out -> d_out slots
  run_stack(stream, d_in, P02, 4, 2, 16, mid, (long long)Bn * 16, 16, bufA,
            bufB, preB, nullptr, hA, OUT + 0 * OS, fOt, fOb);
  run_stack(stream, d_in, P03, 4, 2, 16, mid, (long long)Bn * 16, 16, bufA,
            bufB, preB, hA, hB, OUT + 1 * OS, fOt, fOb);
  run_stack(stream, d_in, P04, 4, 2, 16, mid, (long long)Bn * 16, 16, bufA,
            bufB, preB, hB, hD, OUT + 2 * OS, fOt, fOb);

  // ---- y branch ----
  run_stack(stream, d_in, P10, 1, 1, 25, y, 25, (long long)Tn * 25, bufA, bufB,
            preB, nullptr, hD, bufA, (long long)Bn * 16, 16);
  run_stack(stream, d_in, P11, 3, 1, 16, bufA, (long long)Bn * 16, 16, bufA,
            bufB, preB, nullptr, hD, mid, (long long)Bn * 16, 16);
  run_stack(stream, d_in, P12, 4, 2, 16, mid, (long long)Bn * 16, 16, bufA,
            bufB, preB, nullptr, hA, OUT + 3 * OS, fOt, fOb);
  run_stack(stream, d_in, P13, 4, 2, 16, mid, (long long)Bn * 16, 16, bufA,
            bufB, preB, hA, hB, OUT + 4 * OS, fOt, fOb);
  // y4 reuses p13's params with h3 init (faithful to reference quirk).
  run_stack(stream, d_in, P13, 4, 2, 16, mid, (long long)Bn * 16, 16, bufA,
            bufB, preB, hB, hD, OUT + 5 * OS, fOt, fOb);
}
